// SinkhornKnopp_64244120814138
// MI455X (gfx1250) — compile-verified, matrix operand order checked by reading, not run on hardware
//
#include <hip/hip_runtime.h>
#include <hip/hip_bf16.h>

typedef _Float16 half_t;
typedef __attribute__((ext_vector_type(16))) _Float16 v16h;
typedef __attribute__((ext_vector_type(8)))  _Float16 v8h;
typedef __attribute__((ext_vector_type(4)))  _Float16 v4h;
typedef __attribute__((ext_vector_type(8)))  float    v8f;
typedef __attribute__((ext_vector_type(4)))  float    v4f;

#define NDIM 8192
#define EPS 1e-6f
#define NEG_INV_GAMMA (-10.0f)
#define NUM_STEPS 50

// ---- row pass (WMMA) geometry ----
#define CCH 8                       // column chunks for row pass
#define COLS_PER_CHUNK (NDIM / CCH) // 1024
#define RP_WAVES 4                  // waves per block (128 threads)

// ---- column pass geometry ----
#define CP_BLOCK 256
#define CP_COLS_PER_THREAD 8        // 2048 cols per block -> 4 blocks in x
#define CP_COL_BLOCKS (NDIM / (CP_BLOCK * CP_COLS_PER_THREAD))
#define RPC 128                     // rows per chunk
#define RCH (NDIM / RPC)            // 64 row chunks

// K[i,j] = exp(-10 * C[i,j]) stored as f16 (halves iteration bandwidth)
__global__ void sk_init_K(const float* __restrict__ C, half_t* __restrict__ K) {
    size_t idx = ((size_t)blockIdx.x * blockDim.x + threadIdx.x) * 4;
    v4f c = *(const v4f*)(C + idx);
    v4h h;
#pragma unroll
    for (int i = 0; i < 4; ++i) h[i] = (half_t)__expf(c[i] * NEG_INV_GAMMA);
    *(v4h*)(K + idx) = h;
}

__global__ void sk_init_EV(float* __restrict__ EV) {
    int j = blockIdx.x * blockDim.x + threadIdx.x;
    if (j < NDIM) EV[j] = 2.71828182845904523536f;  // exp(v0), v0 = ones
}

// Row pass: r_part[cc][i] = sum_{j in chunk cc} K[i][j] * EV[j]
// Each wave owns a 16-row tile; per 16x32 tile: W = K_tile * EV (pk_mul in
// memory-index space), then v_wmma_f32_16x16x32_f16(W, ones) -> row sums of W
// (layout-robust: B==ones makes D independent of the hardware K permutation).
__global__ __launch_bounds__(RP_WAVES * 32) void
sk_row_pass(const half_t* __restrict__ K, const float* __restrict__ EV,
            float* __restrict__ r_part) {
    __shared__ __align__(32) half_t evh[COLS_PER_CHUNK];
    const int cc   = blockIdx.y;
    const int col0 = cc * COLS_PER_CHUNK;

    for (int j = threadIdx.x; j < COLS_PER_CHUNK; j += blockDim.x)
        evh[j] = (half_t)EV[col0 + j];
    __syncthreads();

    const int wave = threadIdx.x >> 5;
    const int lane = threadIdx.x & 31;
    const int tileRow = (blockIdx.x * RP_WAVES + wave) * 16;
    const int rowA = tileRow + (lane & 15);
    const int koff = (lane >> 4) * 16;   // lanes 0-15: k 0..15, lanes 16-31: k 16..31

    const half_t* __restrict__ arow = K + (size_t)rowA * NDIM + col0 + koff;
    const half_t* __restrict__ evp  = evh + koff;

    v16h ones;
#pragma unroll
    for (int i = 0; i < 16; ++i) ones[i] = (_Float16)1.0f;

    v8f acc = {};
#pragma unroll 2
    for (int ks = 0; ks < COLS_PER_CHUNK; ks += 32) {
        v16h a = *(const v16h*)(arow + ks);   // 2x global_load_b128
        v16h e = *(const v16h*)(evp + ks);    // 2x ds_load_b128
        v16h w = a * e;                       // 8x v_pk_mul_f16
        acc = __builtin_amdgcn_wmma_f32_16x16x32_f16(
                  false, w, false, ones, (short)0, acc, false, false);
    }

    // f32 C/D layout: acc[r] = D[M = r + 8*(lane>=16)][N = lane&15]; all N equal
    if ((lane & 15) == 0) {
        const int rbase = tileRow + ((lane >> 4) << 3);
        float* __restrict__ dst = r_part + (size_t)cc * NDIM + rbase;
#pragma unroll
        for (int r = 0; r < 8; ++r) dst[r] = acc[r];
    }
}

// EU[i] = (d[i]+eps) / sum_cc r_part[cc][i]
__global__ void sk_reduce_u(const float* __restrict__ r_part,
                            const float* __restrict__ d,
                            float* __restrict__ EU) {
    int i = blockIdx.x * blockDim.x + threadIdx.x;
    if (i >= NDIM) return;
    float s = 0.f;
#pragma unroll
    for (int cc = 0; cc < CCH; ++cc) s += r_part[(size_t)cc * NDIM + i];
    EU[i] = (d[i] + EPS) / s;
}

// Column pass: c_part[rc][j] = sum_{i in chunk rc} K[i][j] * EU[i]
// Coalesced: lanes stride columns (b128 f16 loads), EU broadcast from LDS.
__global__ __launch_bounds__(CP_BLOCK) void
sk_col_pass(const half_t* __restrict__ K, const float* __restrict__ EU,
            float* __restrict__ c_part) {
    __shared__ float eus[RPC];
    const int colBase = blockIdx.x * (CP_BLOCK * CP_COLS_PER_THREAD)
                      + threadIdx.x * CP_COLS_PER_THREAD;
    const int r0 = blockIdx.y * RPC;

    for (int i = threadIdx.x; i < RPC; i += CP_BLOCK) eus[i] = EU[r0 + i];
    __syncthreads();

    float acc[CP_COLS_PER_THREAD] = {};
    const half_t* __restrict__ kp = K + (size_t)r0 * NDIM + colBase;
#pragma unroll 4
    for (int i = 0; i < RPC; ++i) {
        v8h kv = *(const v8h*)kp;
        kp += NDIM;
        const float eu = eus[i];
#pragma unroll
        for (int c = 0; c < CP_COLS_PER_THREAD; ++c)
            acc[c] += (float)kv[c] * eu;   // v_fma_mix / cvt+fma, f32 accum
    }

    float* __restrict__ dst = c_part + (size_t)blockIdx.y * NDIM + colBase;
#pragma unroll
    for (int c = 0; c < CP_COLS_PER_THREAD; ++c) dst[c] = acc[c];
}

// EV[j] = (s[j]+eps) / sum_rc c_part[rc][j]
__global__ void sk_reduce_v(const float* __restrict__ c_part,
                            const float* __restrict__ s,
                            float* __restrict__ EV) {
    int j = blockIdx.x * blockDim.x + threadIdx.x;
    if (j >= NDIM) return;
    float sum = 0.f;
#pragma unroll 8
    for (int rc = 0; rc < RCH; ++rc) sum += c_part[(size_t)rc * NDIM + j];
    EV[j] = (s[j] + EPS) / sum;
}

// P[i][j] = EU[i] * exp(-10*C[i][j]) * EV[j]  (f32 recompute for accuracy)
__global__ void sk_finalize(const float* __restrict__ C,
                            const float* __restrict__ EU,
                            const float* __restrict__ EV,
                            float* __restrict__ out) {
    size_t idx = ((size_t)blockIdx.x * blockDim.x + threadIdx.x) * 4;
    const size_t row = idx >> 13;       // / 8192 (blocks never straddle rows)
    const int j = (int)(idx & (NDIM - 1));
    const float u = EU[row];
    v4f c  = *(const v4f*)(C + idx);
    v4f ev = *(const v4f*)(EV + j);
    v4f o;
#pragma unroll
    for (int i = 0; i < 4; ++i)
        o[i] = u * __expf(c[i] * NEG_INV_GAMMA) * ev[i];
    *(v4f*)(out + idx) = o;
}

extern "C" void kernel_launch(void* const* d_in, const int* in_sizes, int n_in,
                              void* d_out, int out_size, void* d_ws, size_t ws_size,
                              hipStream_t stream) {
    const float* C = (const float*)d_in[0];
    const float* d = (const float*)d_in[1];
    const float* s = (const float*)d_in[2];
    float* out = (float*)d_out;

    // workspace layout
    char* w = (char*)d_ws;
    half_t* K      = (half_t*)w;                               // 128 MB
    float*  r_part = (float*)(w + (size_t)NDIM * NDIM * sizeof(half_t));
    float*  c_part = r_part + (size_t)CCH * NDIM;              // 2 MB
    float*  EU     = c_part + (size_t)RCH * NDIM;
    float*  EV     = EU + NDIM;

    const size_t nelem = (size_t)NDIM * NDIM;

    // K = exp(-C/gamma) in f16; EV = exp(ones)
    sk_init_K<<<dim3((unsigned)(nelem / (256 * 4))), dim3(256), 0, stream>>>(C, K);
    sk_init_EV<<<dim3(NDIM / 256), dim3(256), 0, stream>>>(EV);

    const dim3 rpGrid(NDIM / 16 / RP_WAVES, CCH);       // 128 x 8
    const dim3 rpBlk(RP_WAVES * 32);                    // 128 threads
    const dim3 cpGrid(CP_COL_BLOCKS, RCH);              // 4 x 64
    const dim3 cpBlk(CP_BLOCK);
    const dim3 vGrid(NDIM / 256), vBlk(256);

    for (int it = 0; it < NUM_STEPS; ++it) {
        sk_row_pass<<<rpGrid, rpBlk, 0, stream>>>(K, EV, r_part);
        sk_reduce_u<<<vGrid, vBlk, 0, stream>>>(r_part, d, EU);
        sk_col_pass<<<cpGrid, cpBlk, 0, stream>>>(K, EU, c_part);
        sk_reduce_v<<<vGrid, vBlk, 0, stream>>>(c_part, s, EV);
    }

    sk_finalize<<<dim3((unsigned)(nelem / (256 * 4))), dim3(256), 0, stream>>>(C, EU, EV, out);
}
